// ResidualBlock_30185030157084
// MI455X (gfx1250) — compile-verified, matrix operand order checked by reading, not run on hardware
//
#include <hip/hip_runtime.h>

// ---------------- problem constants (from reference) ----------------
#define B_      8
#define CIN     256
#define COUT    256
#define H_      64
#define W_      64
#define HW      4096          // 64*64
#define T_      512
#define GROUPS  32
#define CPG     8             // channels per group
#define THR     0.001f
#define EPS     1e-5f

// GEMM tiling
#define NB      64            // spatial (N) tile per block  = one full W row
#define COB     128           // output-channel (M) tile per block
#define LDSROW  80            // padded LDS row stride (bytes): 16B aligned, conflict-free
#define KSTEPS  36            // 9 taps * 4 ci-chunks of 64

typedef __attribute__((ext_vector_type(4))) int   v4i;
typedef __attribute__((ext_vector_type(8))) int   v8i;
typedef __attribute__((ext_vector_type(4))) float v4f;

// --- CDNA5 async global->LDS copy (ASYNCcnt-tracked), ISA §15.18.3 op 98 ---
__device__ __forceinline__ unsigned lds_off(const void* p) {
  return (unsigned)(unsigned long long)p;   // LDS aperture: addr[31:0] = LDS byte offset
}
__device__ __forceinline__ void async_b128(unsigned ldsoff, const void* gaddr) {
  asm volatile("global_load_async_to_lds_b128 %0, %1, off"
               :: "v"(ldsoff), "v"((unsigned long long)gaddr) : "memory");
}
__device__ __forceinline__ void wait_async0() {
  asm volatile("s_wait_asynccnt 0x0" ::: "memory");
}

// ---------------- ternarize conv weights: OIHW fp32 -> [k9][co][ci] int8 ----
__global__ __launch_bounds__(256) void k_ternarize(const float* __restrict__ Wsrc,
                                                   signed char* __restrict__ tw) {
  int e = blockIdx.x * 256 + threadIdx.x;        // 9*256*256 elements
  int ci = e & 255;
  int co = (e >> 8) & 255;
  int koff = e >> 16;
  float w = Wsrc[((long)co * CIN + ci) * 9 + koff];
  w = fminf(fmaxf(w, -1.f), 1.f);
  tw[e] = (w > THR) ? (signed char)1 : ((w < -THR) ? (signed char)-1 : (signed char)0);
}

// ---------------- GroupNorm stats, NCHW source (8 contiguous channels) ------
__global__ __launch_bounds__(256) void k_gn_stats_nchw(const float* __restrict__ x,
                                                       float* __restrict__ mu,
                                                       float* __restrict__ rstd) {
  int bg = blockIdx.x;                            // b*32+g, 256 blocks
  const float* base = x + (long)bg * CPG * HW;    // 8 channels contiguous in NCHW
  float s = 0.f, s2 = 0.f;
  for (int i = threadIdx.x; i < CPG * HW; i += 256) {
    float v = base[i];
    s += v; s2 += v * v;
  }
  __shared__ float rs[256], rq[256];
  rs[threadIdx.x] = s; rq[threadIdx.x] = s2;
  __syncthreads();
  for (int o = 128; o > 0; o >>= 1) {
    if (threadIdx.x < o) { rs[threadIdx.x] += rs[threadIdx.x + o]; rq[threadIdx.x] += rq[threadIdx.x + o]; }
    __syncthreads();
  }
  if (threadIdx.x == 0) {
    float m = rs[0] * (1.f / (CPG * HW));
    float var = rq[0] * (1.f / (CPG * HW)) - m * m;
    mu[bg] = m;
    rstd[bg] = rsqrtf(var + EPS);
  }
}

// ---------------- GroupNorm stats, NHWC source ------------------------------
__global__ __launch_bounds__(256) void k_gn_stats_nhwc(const float* __restrict__ y,
                                                       float* __restrict__ mu,
                                                       float* __restrict__ rstd) {
  int bg = blockIdx.x;
  int b = bg >> 5, g = bg & 31;
  const float* base = y + (long)b * HW * COUT + g * CPG;
  float s = 0.f, s2 = 0.f;
  for (int i = threadIdx.x; i < CPG * HW; i += 256) {
    int hw = i >> 3, c = i & 7;
    float v = base[(long)hw * COUT + c];
    s += v; s2 += v * v;
  }
  __shared__ float rs[256], rq[256];
  rs[threadIdx.x] = s; rq[threadIdx.x] = s2;
  __syncthreads();
  for (int o = 128; o > 0; o >>= 1) {
    if (threadIdx.x < o) { rs[threadIdx.x] += rs[threadIdx.x + o]; rq[threadIdx.x] += rq[threadIdx.x + o]; }
    __syncthreads();
  }
  if (threadIdx.x == 0) {
    float m = rs[0] * (1.f / (CPG * HW));
    float var = rq[0] * (1.f / (CPG * HW)) - m * m;
    mu[bg] = m;
    rstd[bg] = rsqrtf(var + EPS);
  }
}

// ---------------- GN + relu + fake_quant -> int8 NHWC, src NCHW -------------
__global__ __launch_bounds__(256) void k_quant_nchw(const float* __restrict__ x,
                                                    const float* __restrict__ mu,
                                                    const float* __restrict__ rstd,
                                                    const float* __restrict__ gamma,
                                                    const float* __restrict__ beta,
                                                    const float* __restrict__ scale,
                                                    signed char* __restrict__ q) {
  int idx = blockIdx.x * 256 + threadIdx.x;       // (b*HW+hw)*256 + c
  int c = idx & 255;
  int p = idx >> 8;
  int b = p >> 12;
  int hw = p & 4095;
  int g = c >> 3;
  float v = x[((long)b * CIN + c) * HW + hw];
  float m = mu[b * GROUPS + g], r = rstd[b * GROUPS + g];
  v = (v - m) * r * gamma[c] + beta[c];
  v = fmaxf(v, 0.f);
  float s = fabsf(scale[0]);
  float t = rintf(v / s);
  t = fminf(fmaxf(t, -1.f), 1.f);
  q[idx] = (signed char)t;
}

// ---------------- GN + relu + fake_quant -> int8 NHWC, src NHWC -------------
__global__ __launch_bounds__(256) void k_quant_nhwc(const float* __restrict__ y,
                                                    const float* __restrict__ mu,
                                                    const float* __restrict__ rstd,
                                                    const float* __restrict__ gamma,
                                                    const float* __restrict__ beta,
                                                    const float* __restrict__ scale,
                                                    signed char* __restrict__ q) {
  int idx = blockIdx.x * 256 + threadIdx.x;
  int c = idx & 255;
  int p = idx >> 8;
  int b = p >> 12;
  int g = c >> 3;
  float v = y[idx];
  float m = mu[b * GROUPS + g], r = rstd[b * GROUPS + g];
  v = (v - m) * r * gamma[c] + beta[c];
  v = fmaxf(v, 0.f);
  float s = fabsf(scale[0]);
  float t = rintf(v / s);
  t = fminf(fmaxf(t, -1.f), 1.f);
  q[idx] = (signed char)t;
}

// ---------------- time conditioning: relu(te) @ tern(Wt)^T + bt -------------
__global__ __launch_bounds__(256) void k_time_mlp(const float* __restrict__ te,
                                                  const float* __restrict__ Wt,
                                                  const float* __restrict__ bt,
                                                  float* __restrict__ tb) {
  int i = blockIdx.x * 256 + threadIdx.x;         // 8*256
  int b = i >> 8, co = i & 255;
  float acc = 0.f;
  for (int t = 0; t < T_; ++t) {
    float a = fmaxf(te[b * T_ + t], 0.f);
    float w = fminf(fmaxf(Wt[co * T_ + t], -1.f), 1.f);
    float tv = (w > THR) ? 1.f : ((w < -THR) ? -1.f : 0.f);
    acc += a * tv;
  }
  tb[i] = acc + bt[co];
}

// load one 16x64 i8 fragment (A or B role) from an LDS tile row-block.
// CDNA5 8-bit fragment packing: vgpr pair (2v,2v+1) = 8 bytes at
// K-offset {0,16,32,48}[v] + (lane>=16)*8; lane row = lane&15.
__device__ __forceinline__ v8i frag_ld(const signed char* rowbase) {
  v8i f;
  unsigned long long d;
  d = *(const unsigned long long*)(rowbase + 0);  f[0] = (int)d; f[1] = (int)(d >> 32);
  d = *(const unsigned long long*)(rowbase + 16); f[2] = (int)d; f[3] = (int)(d >> 32);
  d = *(const unsigned long long*)(rowbase + 32); f[4] = (int)d; f[5] = (int)(d >> 32);
  d = *(const unsigned long long*)(rowbase + 48); f[6] = (int)d; f[7] = (int)(d >> 32);
  return f;
}

// ---------------- ternary conv as int8 WMMA GEMM ----------------------------
// out(n=(b,h,w), co) = sum_{k9, ci} tW[k9][co][ci] * q[shift(n,k9)][ci]
// MODE 0: out NHWC fp32 = |s|*acc + bias[co] + tb[b][co]
// MODE 1: out NCHW fp32 = |s|*acc + bias[co] + xres (residual), final output
template <int MODE>
__global__ __launch_bounds__(256) void k_conv_wmma(const signed char* __restrict__ tW,
                                                   const signed char* __restrict__ q,
                                                   const float* __restrict__ bias,
                                                   const float* __restrict__ tb,
                                                   const float* __restrict__ xres,
                                                   const float* __restrict__ scale,
                                                   float* __restrict__ out) {
  __shared__ signed char sA[2][COB * LDSROW];     // 128 co rows x 64 ci bytes (padded)
  __shared__ signed char sB[2][NB * LDSROW];      // 64 n rows x 64 ci bytes (padded)

  const int tid = threadIdx.x;
  const int lane = tid & 31;
  const int wave = tid >> 5;
  const int waveCo = (wave & 3) * 32;             // 4 co groups of 32
  const int waveN = (wave >> 2) * 32;             // 2 n groups of 32
  const int nBase = blockIdx.x * NB;              // one full W row of one (b,h)
  const int coBase = blockIdx.y * COB;

  // staging coordinates: 4 threads per 64-byte row
  const int srow = tid >> 2;                      // 0..63
  const int chnk = (tid & 3) * 16;                // 16-byte chunk
  const int bb = nBase >> 12;                     // batch of this row-tile
  const int bh = (nBase & 4095) >> 6;             // image row of this tile
  const long qrowbase = ((long)bb * HW) * (long)CIN;

  // stage tiles for k-step `step` into buffers dA/dB using async global->LDS
  auto stage = [&](int step, signed char* dA, signed char* dB) {
    const int koff = step >> 2;                   // tap 0..8
    const int kc = step & 3;                      // ci chunk of 64
    const int kh = koff / 3, kw = koff - kh * 3;
    // A: 128 rows of ternary weights (always in-bounds)
    const long w0 = ((long)(koff * COUT + coBase + srow)) * CIN + kc * 64 + chnk;
    async_b128(lds_off(dA + srow * LDSROW + chnk), tW + w0);
    async_b128(lds_off(dA + (srow + 64) * LDSROW + chnk), tW + w0 + 64L * CIN);
    // B: 64 rows = shifted image row (zero-pad outside)
    const int sh = bh + kh - 1;
    const int sw = srow + kw - 1;
    const bool inb = (sh >= 0) & (sh < H_) & (sw >= 0) & (sw < W_);
    if (inb) {
      const long src = qrowbase + ((long)(sh * W_ + sw)) * CIN + kc * 64 + chnk;
      async_b128(lds_off(dB + srow * LDSROW + chnk), q + src);
    } else {
      *(v4i*)(dB + srow * LDSROW + chnk) = (v4i){0, 0, 0, 0};   // zero pad rows
    }
  };

  v8i acc[2][2] = {};                             // 32x32 per wave (2x2 of 16x16)

  // prologue: fill buffer 0
  stage(0, sA[0], sB[0]);
  wait_async0();
  __syncthreads();

  const int g8 = (lane >> 4) * 8;                 // K sub-range select
  const int rm = lane & 15;                       // fragment row

  for (int s = 0; s < KSTEPS; ++s) {
    const int cur = s & 1;
    // prefetch next tiles into the other buffer while we compute
    if (s + 1 < KSTEPS) stage(s + 1, sA[cur ^ 1], sB[cur ^ 1]);

    const signed char* cA = sA[cur];
    const signed char* cB = sB[cur];
    v8i a0 = frag_ld(cA + (waveCo + rm) * LDSROW + g8);
    v8i a1 = frag_ld(cA + (waveCo + 16 + rm) * LDSROW + g8);
    v8i b0 = frag_ld(cB + (waveN + rm) * LDSROW + g8);
    v8i b1 = frag_ld(cB + (waveN + 16 + rm) * LDSROW + g8);
    acc[0][0] = __builtin_amdgcn_wmma_i32_16x16x64_iu8(true, a0, true, b0, acc[0][0], false, false);
    acc[0][1] = __builtin_amdgcn_wmma_i32_16x16x64_iu8(true, a0, true, b1, acc[0][1], false, false);
    acc[1][0] = __builtin_amdgcn_wmma_i32_16x16x64_iu8(true, a1, true, b0, acc[1][0], false, false);
    acc[1][1] = __builtin_amdgcn_wmma_i32_16x16x64_iu8(true, a1, true, b1, acc[1][1], false, false);

    wait_async0();                                // next buffer fully in LDS
    __syncthreads();                              // all waves done with cur buffer
  }

  // epilogue: lane holds col n = lane&15, rows m = r + (lane>>4)*8, r=0..7
  const float s = fabsf(scale[0]);
#pragma unroll
  for (int cc = 0; cc < 2; ++cc) {
#pragma unroll
    for (int nn = 0; nn < 2; ++nn) {
      const int n = nBase + waveN + nn * 16 + (lane & 15);
      const int ob = n >> 12;
      const int ohw = n & 4095;
      const int mb = coBase + waveCo + cc * 16 + (lane >> 4) * 8;  // 8-aligned
      if (MODE == 0) {
        const v4f bia0 = *(const v4f*)(bias + mb);
        const v4f bia1 = *(const v4f*)(bias + mb + 4);
        const v4f tbb0 = *(const v4f*)(tb + ob * COUT + mb);
        const v4f tbb1 = *(const v4f*)(tb + ob * COUT + mb + 4);
        v4f lo, hi;
#pragma unroll
        for (int r = 0; r < 4; ++r) {
          lo[r] = s * (float)acc[cc][nn][r] + bia0[r] + tbb0[r];
          hi[r] = s * (float)acc[cc][nn][r + 4] + bia1[r] + tbb1[r];
        }
        *(v4f*)(out + (long)n * COUT + mb) = lo;            // NHWC y1
        *(v4f*)(out + (long)n * COUT + mb + 4) = hi;
      } else {
#pragma unroll
        for (int r = 0; r < 8; ++r) {
          const int co = mb + r;
          const long idx = ((long)ob * COUT + co) * HW + ohw;  // NCHW final
          out[idx] = s * (float)acc[cc][nn][r] + bias[co] + xres[idx];
        }
      }
    }
  }
}

// ---------------- host launcher --------------------------------------------
extern "C" void kernel_launch(void* const* d_in, const int* in_sizes, int n_in,
                              void* d_out, int out_size, void* d_ws, size_t ws_size,
                              hipStream_t stream) {
  const float* x     = (const float*)d_in[0];
  const float* temb  = (const float*)d_in[1];
  const float* gn1_g = (const float*)d_in[2];
  const float* gn1_b = (const float*)d_in[3];
  const float* s1    = (const float*)d_in[4];
  const float* W1    = (const float*)d_in[5];
  const float* b1    = (const float*)d_in[6];
  const float* Wt    = (const float*)d_in[7];
  const float* bt    = (const float*)d_in[8];
  const float* gn2_g = (const float*)d_in[9];
  const float* gn2_b = (const float*)d_in[10];
  const float* s2    = (const float*)d_in[11];
  const float* W2    = (const float*)d_in[12];
  const float* b2    = (const float*)d_in[13];
  float* out = (float*)d_out;

  // workspace carve-up (256B aligned)
  char* ws = (char*)d_ws;
  size_t off = 0;
  auto carve = [&](size_t bytes) -> char* {
    char* p = ws + off;
    off += (bytes + 255) & ~(size_t)255;
    return p;
  };
  signed char* q1  = (signed char*)carve((size_t)B_ * HW * CIN);        // 8 MB
  signed char* q2  = (signed char*)carve((size_t)B_ * HW * COUT);       // 8 MB
  float*       y1  = (float*)carve((size_t)B_ * HW * COUT * 4);         // 32 MB
  signed char* tW1 = (signed char*)carve((size_t)9 * COUT * CIN);       // 576 KB
  signed char* tW2 = (signed char*)carve((size_t)9 * COUT * COUT);      // 576 KB
  float*       mu1 = (float*)carve(B_ * GROUPS * 4);
  float*       rs1 = (float*)carve(B_ * GROUPS * 4);
  float*       mu2 = (float*)carve(B_ * GROUPS * 4);
  float*       rs2 = (float*)carve(B_ * GROUPS * 4);
  float*       tb  = (float*)carve((size_t)B_ * COUT * 4);
  (void)ws_size; (void)in_sizes; (void)n_in; (void)out_size;

  const int elems = B_ * HW * CIN;                 // 8,388,608

  // 1) ternarize weights
  k_ternarize<<<(9 * COUT * CIN) / 256, 256, 0, stream>>>(W1, tW1);
  k_ternarize<<<(9 * COUT * COUT) / 256, 256, 0, stream>>>(W2, tW2);
  // 2) GN1 stats + time MLP
  k_gn_stats_nchw<<<B_ * GROUPS, 256, 0, stream>>>(x, mu1, rs1);
  k_time_mlp<<<(B_ * COUT) / 256, 256, 0, stream>>>(temb, Wt, bt, tb);
  // 3) quantize act1 -> NHWC int8
  k_quant_nchw<<<elems / 256, 256, 0, stream>>>(x, mu1, rs1, gn1_g, gn1_b, s1, q1);
  // 4) conv1 (WMMA iu8, async double-buffered) -> y1 NHWC fp32 (+ bias + time)
  dim3 ggrid(B_ * HW / NB, COUT / COB);            // (512, 2)
  k_conv_wmma<0><<<ggrid, 256, 0, stream>>>(tW1, q1, b1, tb, nullptr, s1, y1);
  // 5) GN2 stats + quantize act2
  k_gn_stats_nhwc<<<B_ * GROUPS, 256, 0, stream>>>(y1, mu2, rs2);
  k_quant_nhwc<<<elems / 256, 256, 0, stream>>>(y1, mu2, rs2, gn2_g, gn2_b, s2, q2);
  // 6) conv2 (WMMA iu8) -> NCHW out (+ bias + residual x)
  k_conv_wmma<1><<<ggrid, 256, 0, stream>>>(tW2, q2, b2, nullptr, x, s2, out);
}
